// GRUObservationCellLogvar3_59889023976085
// MI455X (gfx1250) — matrix operand order repaired and checked by reading, not compiled, not run on hardware
//
#include <hip/hip_runtime.h>
#include <hip/hip_bf16.h>
#include <math.h>
#include <stdint.h>

typedef _Float16 f16;
typedef __attribute__((ext_vector_type(16))) _Float16 v16h;
typedef __attribute__((ext_vector_type(8)))  _Float16 v8h;
typedef __attribute__((ext_vector_type(8)))  float    v8f;

#define F_DIM   64
#define H_DIM   256
#define NOBS    50000
#define NTOT    100000
#define ROWS    32            // obs rows per block (2 WMMA M-tiles)
#define KA      1280          // 1024 (x) + 256 (h) concatenated K for gate GEMM
#define KT      40            // k-steps (KA/32)
#define SA      1288          // LDS A row stride (f16 elems), multiple of 8
#define TWO_LOG_LIK_C 1.8378770664093453f

// f16 workspace layout (element offsets)
#define WG1_OFF 0             // 1920 tiles x 512 (tile-packed 768x1280)
#define WG2_OFF 983040
#define ADJ_OFF 1966080       // 64 x 64 row-major
#define MW1_OFF 1970176       // 32 x 512 row-major
#define MW2_OFF 1986560       // 32 x 32
#define MW3_OFF 1987584       // 256 x 32

// ---------------- global (addrspace 1) vector load ----------------
typedef __attribute__((address_space(1))) const v8h* as1_v8h_p;
__device__ __forceinline__ v8h gload_v8h(const f16* p) {
    return *reinterpret_cast<as1_v8h_p>(reinterpret_cast<uintptr_t>(p));
}

// ---------------- WMMA fragment helpers ----------------
// A 16x32 f16: lane<16 row M=lane holds K {0..15},
// lane>=16 row M=lane-16 holds K {16..31}.  Row-major f16 in LDS.
__device__ __forceinline__ v16h lds_load_a(const f16* base, int stride, int kbase, int lane) {
    int m    = lane & 15;
    int koff = (lane & 16) ? 16 : 0;
    const f16* p = base + m * stride + kbase + koff;
    v8h lo = *(const v8h*)p;
    v8h hi = *(const v8h*)(p + 8);
    v16h a;
#pragma unroll
    for (int i = 0; i < 8; ++i) { a[i] = lo[i]; a[i + 8] = hi[i]; }
    return a;
}

// B 32x16 f16 from tile-packed weights: fragment block = j*KT + t, 1KB per block,
// lane L owns 16 consecutive f16 at block*512 + L*16 (fully coalesced).
__device__ __forceinline__ v16h load_b_packed(const f16* base, int j, int t, int lane) {
    const f16* p = base + (((j * KT) + t) << 9) + (lane << 4);
    v8h lo = gload_v8h(p);
    v8h hi = gload_v8h(p + 8);
    v16h b;
#pragma unroll
    for (int i = 0; i < 8; ++i) { b[i] = lo[i]; b[i + 8] = hi[i]; }
    return b;
}

// B 32x16 f16 from row-major W (small matrices): lane = column N (mod16),
// lane<16 K 0..15, lane>=16 K 16..31; B[k][n] = W[n][k].
__device__ __forceinline__ v16h glb_load_b(const f16* W, int rowStride, int colbase,
                                           int kbase, int lane) {
    int n  = lane & 15;
    int kb = (lane & 16) ? 16 : 0;
    const f16* p = W + (size_t)(colbase + n) * rowStride + kbase + kb;
    v8h lo = gload_v8h(p);
    v8h hi = gload_v8h(p + 8);
    v16h b;
#pragma unroll
    for (int i = 0; i < 8; ++i) { b[i] = lo[i]; b[i + 8] = hi[i]; }
    return b;
}

__device__ __forceinline__ v8f wmma16(v16h a, v16h b, v8f c) {
    return __builtin_amdgcn_wmma_f32_16x16x32_f16(false, a, false, b, (short)0, c,
                                                  false, false);
}

__device__ __forceinline__ float sigmoidf_(float x) { return 1.f / (1.f + expf(-x)); }

// ---------------- GRU gate GEMM: [x|h](32x1280) x Wg^T -> 32x768 ----------------
// Software-pipelined: fragments for k-step t+1 are loaded while the 8 WMMAs of
// step t issue; B shared by 2 M-tiles, A shared by 4 N-tiles.
__device__ void gates_gemm(const f16* __restrict__ Wp, const f16* __restrict__ sA_,
                           float* __restrict__ sGrz, float* __restrict__ sGni,
                           float* __restrict__ sGnh,
                           const float* __restrict__ b_ih, const float* __restrict__ b_hh,
                           int wave, int lane) {
    int n    = lane & 15;
    int moff = (lane & 16) ? 8 : 0;

    // ---- r/z tiles: j in {wave, wave+8, wave+16, wave+24}, full K=1280 ----
    {
        v8f acc[4][2];
#pragma unroll
        for (int jj = 0; jj < 4; ++jj) { acc[jj][0] = (v8f){}; acc[jj][1] = (v8f){}; }
        v16h a0 = lds_load_a(sA_,           SA, 0, lane);
        v16h a1 = lds_load_a(sA_ + 16 * SA, SA, 0, lane);
        v16h b[4];
#pragma unroll
        for (int jj = 0; jj < 4; ++jj) b[jj] = load_b_packed(Wp, wave + 8 * jj, 0, lane);
        for (int t = 0; t < KT - 1; ++t) {
            v16h na0 = lds_load_a(sA_,           SA, (t + 1) * 32, lane);
            v16h na1 = lds_load_a(sA_ + 16 * SA, SA, (t + 1) * 32, lane);
            v16h nb[4];
#pragma unroll
            for (int jj = 0; jj < 4; ++jj)
                nb[jj] = load_b_packed(Wp, wave + 8 * jj, t + 1, lane);
#pragma unroll
            for (int jj = 0; jj < 4; ++jj) {
                acc[jj][0] = wmma16(a0, b[jj], acc[jj][0]);
                acc[jj][1] = wmma16(a1, b[jj], acc[jj][1]);
            }
            a0 = na0; a1 = na1;
#pragma unroll
            for (int jj = 0; jj < 4; ++jj) b[jj] = nb[jj];
        }
#pragma unroll
        for (int jj = 0; jj < 4; ++jj) {
            acc[jj][0] = wmma16(a0, b[jj], acc[jj][0]);
            acc[jj][1] = wmma16(a1, b[jj], acc[jj][1]);
        }
#pragma unroll
        for (int jj = 0; jj < 4; ++jj) {
            int colbase = (wave + 8 * jj) * 16;
            float bv = b_ih[colbase + n] + b_hh[colbase + n];
#pragma unroll
            for (int mt = 0; mt < 2; ++mt)
#pragma unroll
                for (int r = 0; r < 8; ++r)
                    sGrz[(mt * 16 + moff + r) * 512 + colbase + n] = acc[jj][mt][r] + bv;
        }
    }

    // ---- n-gate tiles: j in {wave+32, wave+40}; split K at 1024 ----
    {
        v8f acc[2][2];
#pragma unroll
        for (int jj = 0; jj < 2; ++jj) { acc[jj][0] = (v8f){}; acc[jj][1] = (v8f){}; }
        v16h a0 = lds_load_a(sA_,           SA, 0, lane);
        v16h a1 = lds_load_a(sA_ + 16 * SA, SA, 0, lane);
        v16h b[2];
#pragma unroll
        for (int jj = 0; jj < 2; ++jj) b[jj] = load_b_packed(Wp, wave + 32 + 8 * jj, 0, lane);
        for (int t = 0; t < 31; ++t) {            // K 0..1023 : x * W_ih(n)
            v16h na0 = lds_load_a(sA_,           SA, (t + 1) * 32, lane);
            v16h na1 = lds_load_a(sA_ + 16 * SA, SA, (t + 1) * 32, lane);
            v16h nb[2];
#pragma unroll
            for (int jj = 0; jj < 2; ++jj)
                nb[jj] = load_b_packed(Wp, wave + 32 + 8 * jj, t + 1, lane);
#pragma unroll
            for (int jj = 0; jj < 2; ++jj) {
                acc[jj][0] = wmma16(a0, b[jj], acc[jj][0]);
                acc[jj][1] = wmma16(a1, b[jj], acc[jj][1]);
            }
            a0 = na0; a1 = na1; b[0] = nb[0]; b[1] = nb[1];
        }
#pragma unroll
        for (int jj = 0; jj < 2; ++jj) {
            acc[jj][0] = wmma16(a0, b[jj], acc[jj][0]);
            acc[jj][1] = wmma16(a1, b[jj], acc[jj][1]);
        }
#pragma unroll
        for (int jj = 0; jj < 2; ++jj) {
            int colbase = (wave + 32 + 8 * jj) * 16;
            float bi = b_ih[colbase + n];
#pragma unroll
            for (int mt = 0; mt < 2; ++mt)
#pragma unroll
                for (int r = 0; r < 8; ++r)
                    sGni[(mt * 16 + moff + r) * 256 + (colbase - 512) + n] =
                        acc[jj][mt][r] + bi;
        }
        // K 1024..1279 : h * W_hh(n)  (8 steps, simple loop)
#pragma unroll
        for (int jj = 0; jj < 2; ++jj) { acc[jj][0] = (v8f){}; acc[jj][1] = (v8f){}; }
        for (int t = 32; t < KT; ++t) {
            v16h a0s = lds_load_a(sA_,           SA, t * 32, lane);
            v16h a1s = lds_load_a(sA_ + 16 * SA, SA, t * 32, lane);
#pragma unroll
            for (int jj = 0; jj < 2; ++jj) {
                v16h bb = load_b_packed(Wp, wave + 32 + 8 * jj, t, lane);
                acc[jj][0] = wmma16(a0s, bb, acc[jj][0]);
                acc[jj][1] = wmma16(a1s, bb, acc[jj][1]);
            }
        }
#pragma unroll
        for (int jj = 0; jj < 2; ++jj) {
            int colbase = (wave + 32 + 8 * jj) * 16;
            float bh = b_hh[colbase + n];
#pragma unroll
            for (int mt = 0; mt < 2; ++mt)
#pragma unroll
                for (int r = 0; r < 8; ++r)
                    sGnh[(mt * 16 + moff + r) * 256 + (colbase - 512) + n] =
                        acc[jj][mt][r] + bh;
        }
    }
}

// ---------------- helper kernels ----------------
// Tile-packed gate weights: block = j*KT + t holds the 32x16 B fragment for
// columns j*16..j*16+15, K t*32..t*32+31, in exact lane order (lane L: 16 f16).
extern "C" __global__ void pack_gates_tiled_kernel(const float* __restrict__ w_ih,
                                                   const float* __restrict__ w_hh,
                                                   f16* __restrict__ dst) {
    int total = 48 * KT * 512;
    for (int idx = blockIdx.x * blockDim.x + threadIdx.x; idx < total;
         idx += gridDim.x * blockDim.x) {
        int blockId = idx >> 9;
        int within  = idx & 511;
        int lane = within >> 4;
        int i    = within & 15;
        int j = blockId / KT, t = blockId - j * KT;
        int col = j * 16 + (lane & 15);
        int k = t * 32 + ((lane & 16) ? 16 : 0) + i;
        float v = (k < 1024) ? w_ih[col * 1024 + k] : w_hh[col * 256 + (k - 1024)];
        dst[idx] = (f16)v;
    }
}

extern "C" __global__ void cvt_f32_f16_kernel(const float* __restrict__ src,
                                              f16* __restrict__ dst, int n) {
    for (int idx = blockIdx.x * blockDim.x + threadIdx.x; idx < n;
         idx += gridDim.x * blockDim.x)
        dst[idx] = (f16)src[idx];
}

extern "C" __global__ void copy_h_kernel(const float4* __restrict__ src,
                                         float4* __restrict__ dst, int n4) {
    for (int idx = blockIdx.x * blockDim.x + threadIdx.x; idx < n4;
         idx += gridDim.x * blockDim.x)
        dst[idx] = src[idx];
}

// ---------------- fused main kernel: 32 obs rows / block, 256 threads ----------------
extern "C" __global__ __launch_bounds__(256)
void gru_obs_fused_kernel(const float* __restrict__ h,
                          const float* __restrict__ p_obs,
                          const float* __restrict__ X_obs,
                          const float* __restrict__ M_obs,
                          const int*   __restrict__ i_obs,
                          const float* __restrict__ w_prep,
                          const float* __restrict__ w_prep2,
                          const float* __restrict__ bias_prep,
                          const float* __restrict__ g1_b_ih, const float* __restrict__ g1_b_hh,
                          const float* __restrict__ g2_b_ih, const float* __restrict__ g2_b_hh,
                          const float* __restrict__ gcn_w1, const float* __restrict__ gcn_b1,
                          const float* __restrict__ gcn_w2, const float* __restrict__ gcn_b2,
                          const float* __restrict__ m_b1, const float* __restrict__ m_b2,
                          const float* __restrict__ m_b3,
                          const f16*   __restrict__ ws16,
                          float* __restrict__ out_h,
                          float* __restrict__ out_loss) {
    __shared__ __align__(16) f16  sA[ROWS * SA];     // [x|h] f16       (82432 B)
    __shared__ float sGrz[ROWS * 512];               // r,z gate sums   (65536 B)
    __shared__ float sGni[ROWS * 256];               // n gate (ih)     (32768 B)
    __shared__ float sGnh[ROWS * 256];               // n gate (hh)     (32768 B)
    __shared__ float sXv[ROWS * 64], sMean[ROWS * 64], sLogv[ROWS * 64],
                     sErr[ROWS * 64], sM[ROWS * 64]; //                 (40960 B)
    __shared__ float sHobs[ROWS * 64];               // GCN output      ( 8192 B)
    __shared__ __align__(16) f16 sXf16[ROWS * 80];   // X f16 (A-mat)   ( 5120 B)
    __shared__ __align__(16) f16 sS16[ROWS * 80];    // phi(ax) f16     ( 5120 B)
    __shared__ __align__(16) f16 sZ[ROWS * 528];     // [temp|temp2]    (33792 B)
    __shared__ __align__(16) f16 sY1[ROWS * 48];     //                 ( 3072 B)
    __shared__ __align__(16) f16 sY2[ROWS * 48];     //                 ( 3072 B)
    __shared__ int   sIdx[ROWS];
    __shared__ float sW1[32], sB1[32], sW2[32];      // total ~313 KB (<320 KB WGP)

    const int tid  = threadIdx.x;
    const int lane = tid & 31;
    const int wave = tid >> 5;
    const int row0 = blockIdx.x * ROWS;

    const f16* Wg1   = ws16 + WG1_OFF;
    const f16* Wg2   = ws16 + WG2_OFF;
    const f16* Adj16 = ws16 + ADJ_OFF;
    const f16* Mw1   = ws16 + MW1_OFF;
    const f16* Mw2   = ws16 + MW2_OFF;
    const f16* Mw3   = ws16 + MW3_OFF;

    const int n_    = lane & 15;
    const int moff  = (lane & 16) ? 8 : 0;

    // ---- phase 0: obs rows, losses, indices, GCN params ----
    if (tid < ROWS) sIdx[tid] = i_obs[min(row0 + tid, NOBS - 1)];
    if (tid < 32) { sW1[tid] = gcn_w1[tid]; sB1[tid] = gcn_b1[tid]; sW2[tid] = gcn_w2[tid]; }
    for (int idx = tid; idx < ROWS * 64; idx += 256) {
        int r = idx >> 6, f = idx & 63;
        size_t g = (size_t)min(row0 + r, NOBS - 1);
        float mean = p_obs[g * 128 + f];
        float logv = p_obs[g * 128 + 64 + f];
        float x    = X_obs[g * 64 + f];
        float m    = M_obs[g * 64 + f];
        float err  = (x - mean) / expf(0.5f * logv);
        sXv[idx] = x; sMean[idx] = mean; sLogv[idx] = logv; sErr[idx] = err; sM[idx] = m;
        sXf16[r * 80 + f] = (f16)x;
        if (row0 + r < NOBS)
            out_loss[g * 64 + f] = 0.5f * ((err * err + logv + TWO_LOG_LIK_C) * m);
    }
    __syncthreads();

    // ---- gather h rows (f16 at A col 1024) ----
    for (int idx = tid; idx < ROWS * 256; idx += 256) {
        int r = idx >> 8, c = idx & 255;
        sA[r * SA + 1024 + c] = (f16)h[(size_t)sIdx[r] * 256 + c];
    }
    // ---- prep1: x1 = relu(feats . w_prep + bias) * M ----
    for (int pair = tid; pair < ROWS * 64; pair += 256) {
        int r = pair >> 6, f = pair & 63;
        float fe0 = sXv[pair], fe1 = sMean[pair], fe2 = sLogv[pair], fe3 = sErr[pair];
        float m = sM[pair];
        const float* wp = w_prep + f * 64;
        const float* bp = bias_prep + f * 16;
#pragma unroll
        for (int p = 0; p < 16; ++p) {
            float v = bp[p] + fe0 * wp[p] + fe1 * wp[16 + p] + fe2 * wp[32 + p] +
                      fe3 * wp[48 + p];
            sA[r * SA + f * 16 + p] = (f16)(fmaxf(v, 0.f) * m);
        }
    }
    __syncthreads();

    // ---- GRU1 gate GEMM (WMMA) ----
    gates_gemm(Wg1, sA, sGrz, sGni, sGnh, g1_b_ih, g1_b_hh, wave, lane);
    __syncthreads();

    // ---- GRU1 elementwise -> temp (sZ[:,0:256]) ----
    for (int idx = tid; idx < ROWS * 256; idx += 256) {
        int r = idx >> 8, c = idx & 255;
        float rr = sigmoidf_(sGrz[r * 512 + c]);
        float zz = sigmoidf_(sGrz[r * 512 + 256 + c]);
        float nn = tanhf(sGni[idx] + rr * sGnh[idx]);
        float hv = (float)sA[r * SA + 1024 + c];
        sZ[r * 528 + c] = (f16)((1.f - zz) * nn + zz * hv);
    }
    __syncthreads();

    // ---- GCN stage 1: ax = X.adjT (WMMA) then phi(ax) -> sS16 ----
    {
        int mt = wave >> 2, jt = wave & 3, colbase = jt * 16;
        v8f acc = {};
        for (int t = 0; t < 2; ++t) {
            v16h a = lds_load_a(sXf16 + mt * 16 * 80, 80, t * 32, lane);
            v16h b = glb_load_b(Adj16, 64, colbase, t * 32, lane);
            acc = wmma16(a, b, acc);
        }
#pragma unroll
        for (int r = 0; r < 8; ++r) {
            float a0 = acc[r], s = 0.f;
#pragma unroll
            for (int c = 0; c < 32; ++c)
                s += sW2[c] * fmaxf(a0 * sW1[c] + sB1[c], 0.f);
            sS16[(mt * 16 + moff + r) * 80 + colbase + n_] = (f16)s;
        }
    }
    __syncthreads();
    // ---- GCN stage 2: H_obs = s.adjT + b2 (WMMA) ----
    {
        int mt = wave >> 2, jt = wave & 3, colbase = jt * 16;
        v8f acc = {};
        for (int t = 0; t < 2; ++t) {
            v16h a = lds_load_a(sS16 + mt * 16 * 80, 80, t * 32, lane);
            v16h b = glb_load_b(Adj16, 64, colbase, t * 32, lane);
            acc = wmma16(a, b, acc);
        }
        float b2 = gcn_b2[0];
#pragma unroll
        for (int r = 0; r < 8; ++r)
            sHobs[(mt * 16 + moff + r) * 64 + colbase + n_] = acc[r] + b2;
    }
    __syncthreads();

    // ---- prep2 (5 features incl. H_obs) ----
    for (int pair = tid; pair < ROWS * 64; pair += 256) {
        int r = pair >> 6, f = pair & 63;
        float fe0 = sXv[pair], fe1 = sHobs[pair], fe2 = sMean[pair],
              fe3 = sLogv[pair], fe4 = sErr[pair];
        float m = sM[pair];
        const float* wp = w_prep2 + f * 80;
        const float* bp = bias_prep + f * 16;
#pragma unroll
        for (int p = 0; p < 16; ++p) {
            float v = bp[p] + fe0 * wp[p] + fe1 * wp[16 + p] + fe2 * wp[32 + p] +
                      fe3 * wp[48 + p] + fe4 * wp[64 + p];
            sA[r * SA + f * 16 + p] = (f16)(fmaxf(v, 0.f) * m);
        }
    }
    __syncthreads();

    // ---- GRU2 gate GEMM + elementwise -> temp2 (sZ[:,256:512]) ----
    gates_gemm(Wg2, sA, sGrz, sGni, sGnh, g2_b_ih, g2_b_hh, wave, lane);
    __syncthreads();
    for (int idx = tid; idx < ROWS * 256; idx += 256) {
        int r = idx >> 8, c = idx & 255;
        float rr = sigmoidf_(sGrz[r * 512 + c]);
        float zz = sigmoidf_(sGrz[r * 512 + 256 + c]);
        float nn = tanhf(sGni[idx] + rr * sGnh[idx]);
        float hv = (float)sA[r * SA + 1024 + c];
        sZ[r * 528 + 256 + c] = (f16)((1.f - zz) * nn + zz * hv);
    }
    __syncthreads();

    // ---- MLP layer1: relu(z . m_w1^T + b1), K=512, N=32 (WMMA) ----
    if (wave < 4) {
        int mt = wave >> 1, jt = wave & 1, colbase = jt * 16;
        v8f acc = {};
        for (int t = 0; t < 16; ++t) {
            v16h a = lds_load_a(sZ + mt * 16 * 528, 528, t * 32, lane);
            v16h b = glb_load_b(Mw1, 512, colbase, t * 32, lane);
            acc = wmma16(a, b, acc);
        }
        float bv = m_b1[colbase + n_];
#pragma unroll
        for (int r = 0; r < 8; ++r)
            sY1[(mt * 16 + moff + r) * 48 + colbase + n_] = (f16)fmaxf(acc[r] + bv, 0.f);
    }
    __syncthreads();
    // ---- MLP layer2: relu(y1 . m_w2^T + b2), K=32, N=32 (WMMA) ----
    if (wave < 4) {
        int mt = wave >> 1, jt = wave & 1, colbase = jt * 16;
        v16h a = lds_load_a(sY1 + mt * 16 * 48, 48, 0, lane);
        v16h b = glb_load_b(Mw2, 32, colbase, 0, lane);
        v8f acc = {};
        acc = wmma16(a, b, acc);
        float bv = m_b2[colbase + n_];
#pragma unroll
        for (int r = 0; r < 8; ++r)
            sY2[(mt * 16 + moff + r) * 48 + colbase + n_] = (f16)fmaxf(acc[r] + bv, 0.f);
    }
    __syncthreads();
    // ---- MLP layer3: silu(y2 . m_w3^T + b3), K=32, N=256 -> scatter to out_h ----
#pragma unroll
    for (int i = 0; i < 4; ++i) {
        int job = wave * 4 + i;
        int jt = job >> 1, mt = job & 1, colbase = jt * 16;
        v16h a = lds_load_a(sY2 + mt * 16 * 48, 48, 0, lane);
        v16h b = glb_load_b(Mw3, 32, colbase, 0, lane);
        v8f acc = {};
        acc = wmma16(a, b, acc);
        float bv = m_b3[colbase + n_];
#pragma unroll
        for (int r = 0; r < 8; ++r) {
            int mrow = mt * 16 + moff + r;
            if (row0 + mrow < NOBS) {
                float v = acc[r] + bv;
                out_h[(size_t)sIdx[mrow] * 256 + colbase + n_] = v * sigmoidf_(v);
            }
        }
    }
}

// ---------------- launch ----------------
extern "C" void kernel_launch(void* const* d_in, const int* in_sizes, int n_in,
                              void* d_out, int out_size, void* d_ws, size_t ws_size,
                              hipStream_t stream) {
    const float* h        = (const float*)d_in[0];
    const float* p_obs    = (const float*)d_in[1];
    const float* X_obs    = (const float*)d_in[2];
    const float* M_obs    = (const float*)d_in[3];
    const float* adj      = (const float*)d_in[4];
    const int*   i_obs    = (const int*)d_in[5];
    const float* w_prep   = (const float*)d_in[6];
    const float* w_prep2  = (const float*)d_in[7];
    const float* bias_prep= (const float*)d_in[8];
    const float* g1_w_ih  = (const float*)d_in[9];
    const float* g1_w_hh  = (const float*)d_in[10];
    const float* g1_b_ih  = (const float*)d_in[11];
    const float* g1_b_hh  = (const float*)d_in[12];
    const float* g2_w_ih  = (const float*)d_in[13];
    const float* g2_w_hh  = (const float*)d_in[14];
    const float* g2_b_ih  = (const float*)d_in[15];
    const float* g2_b_hh  = (const float*)d_in[16];
    const float* gcn_w1   = (const float*)d_in[17];
    const float* gcn_b1   = (const float*)d_in[18];
    const float* gcn_w2   = (const float*)d_in[19];
    const float* gcn_b2   = (const float*)d_in[20];
    const float* m_w1     = (const float*)d_in[21];
    const float* m_b1     = (const float*)d_in[22];
    const float* m_w2     = (const float*)d_in[23];
    const float* m_b2     = (const float*)d_in[24];
    const float* m_w3     = (const float*)d_in[25];
    const float* m_b3     = (const float*)d_in[26];

    f16*   ws16     = (f16*)d_ws;
    float* out_h    = (float*)d_out;
    float* out_loss = out_h + (size_t)NTOT * H_DIM;

    // per-launch f16 weight packing (cheap: ~8 MB read)
    pack_gates_tiled_kernel<<<960, 256, 0, stream>>>(g1_w_ih, g1_w_hh, ws16 + WG1_OFF);
    pack_gates_tiled_kernel<<<960, 256, 0, stream>>>(g2_w_ih, g2_w_hh, ws16 + WG2_OFF);
    cvt_f32_f16_kernel<<<16, 256, 0, stream>>>(adj,  ws16 + ADJ_OFF, 64 * 64);
    cvt_f32_f16_kernel<<<64, 256, 0, stream>>>(m_w1, ws16 + MW1_OFF, 32 * 512);
    cvt_f32_f16_kernel<<<4,  256, 0, stream>>>(m_w2, ws16 + MW2_OFF, 32 * 32);
    cvt_f32_f16_kernel<<<32, 256, 0, stream>>>(m_w3, ws16 + MW3_OFF, 256 * 32);

    // h_new = copy(h); fused kernel then scatters updated rows (stream-ordered)
    copy_h_kernel<<<4096, 256, 0, stream>>>((const float4*)h, (float4*)out_h,
                                            NTOT * H_DIM / 4);

    gru_obs_fused_kernel<<<(NOBS + ROWS - 1) / ROWS, 256, 0, stream>>>(
        h, p_obs, X_obs, M_obs, i_obs, w_prep, w_prep2, bias_prep,
        g1_b_ih, g1_b_hh, g2_b_ih, g2_b_hh,
        gcn_w1, gcn_b1, gcn_w2, gcn_b2,
        m_b1, m_b2, m_b3, ws16, out_h, out_loss);
}
